// MulitHeadAttention_61572651155727
// MI455X (gfx1250) — compile-verified, matrix operand order checked by reading, not run on hardware
//
#include <hip/hip_runtime.h>
#include <hip/hip_bf16.h>
#include <stdint.h>

// ---------------------------------------------------------------------------
// Multi-head attention forward for MI455X (gfx1250).
// bf16 WMMA + async global->LDS double-buffered staging (CDNA5 ASYNCcnt path).
// GEMMs: 64x256 WG tile, 32x64 per wave (8 WMMAs / 2 chains per K-chunk).
//   x [2,2048,1024] f32, w_qkv [3072,1024], b_qkv [3072],
//   w_proj [1024,1024], b_proj [1024]  ->  out [2,2048,1024] f32
// ---------------------------------------------------------------------------

#define DIM      1024
#define SEQ      2048
#define NBATCH   2
#define NHEADS   16
#define HEADD    64
#define BH       (NBATCH * NHEADS)   // 32
#define MROWS    (NBATCH * SEQ)      // 4096

typedef __attribute__((ext_vector_type(16))) __bf16 v16bf;
typedef __attribute__((ext_vector_type(8)))  float  v8f;

union FragBF {
    v16bf v;
    uint4 q[2];
};

__device__ __forceinline__ __bf16 f2bf(float x) {
    unsigned u = __builtin_bit_cast(unsigned, x);
    u += 0x7fffu + ((u >> 16) & 1u);            // round-to-nearest-even
    unsigned short h = (unsigned short)(u >> 16);
    return __builtin_bit_cast(__bf16, h);
}

// low 32 bits of a flat shared-space pointer are the LDS byte address
__device__ __forceinline__ unsigned lds_off(const void* p) {
    return (unsigned)(unsigned long long)p;
}

// One per-lane async copy: 16 B global -> LDS, tracked by ASYNCcnt.
// INST_OFFSET applies to both global and LDS addresses (ISA 08 §4.4).
#define ASYNC_CP16(ldsdst, gsrc, OFF)                                          \
    asm volatile("global_load_async_to_lds_b128 %0, %1, off offset:" #OFF     \
                 :: "v"(ldsdst), "v"(gsrc) : "memory")

__device__ __forceinline__ void wait_async0() {
    asm volatile("s_wait_asynccnt 0" ::: "memory");
}

// 16x32 A-style / 32x16 B-style bf16 WMMA fragment from a row-major matrix.
// Lane holds row (lane&15); lanes 0-15 take K {0..7,16..23}, lanes 16-31 take
// K {8..15,24..31}: two contiguous 16 B chunks at col+kh / col+16+kh.
__device__ __forceinline__ v16bf load_frag(const __bf16* base, int ld,
                                           int row, int col, int lane) {
    int r  = row + (lane & 15);
    int kh = (lane >> 4) << 3;
    const __bf16* p = base + (size_t)r * ld + col;
    FragBF f;
    f.q[0] = *(const uint4*)(p + kh);
    f.q[1] = *(const uint4*)(p + 16 + kh);
    return f.v;
}

__device__ __forceinline__ v8f wmma_bf16(v16bf a, v16bf b, v8f c) {
    return __builtin_amdgcn_wmma_f32_16x16x32_bf16(
        false, a, false, b, (short)0, c, false, false);
}

// ---------------------------------------------------------------------------
// Kernel 0: f32 -> bf16 conversion
// ---------------------------------------------------------------------------
__global__ void cvt_f32_to_bf16(const float* __restrict__ in,
                                __bf16* __restrict__ out, int n) {
    int i = blockIdx.x * blockDim.x + threadIdx.x;
    if (i < n) out[i] = f2bf(in[i]);
}

// ---------------------------------------------------------------------------
// GEMM core: C[64 x 256] per workgroup, K staged 64-deep, async double buffer.
// 8 waves: wave&1 = 32-row half, wave>>1 = 64-col group; each wave 32x64.
// LDS = 80 KB.
// ---------------------------------------------------------------------------
struct GemmLds {
    __align__(16) __bf16 A[2][64][64];    // 16 KB
    __align__(16) __bf16 B[2][256][64];   // 64 KB
};

__device__ __forceinline__ void gemm_stage(GemmLds* s, int buf,
                                           const __bf16* X, const __bf16* W,
                                           int mbase0, int nbase0, int k,
                                           int t) {
    // B tile 256x64: thread t copies row t (128 B = 8 chunks)
    {
        unsigned d = lds_off(&s->B[buf][t][0]);
        const __bf16* g = W + (size_t)(nbase0 + t) * DIM + k;
        ASYNC_CP16(d, g, 0);   ASYNC_CP16(d, g, 16);
        ASYNC_CP16(d, g, 32);  ASYNC_CP16(d, g, 48);
        ASYNC_CP16(d, g, 64);  ASYNC_CP16(d, g, 80);
        ASYNC_CP16(d, g, 96);  ASYNC_CP16(d, g, 112);
    }
    // A tile 64x64: 512 chunks, 2 per thread (row = t>>2, chunks 2*(t&3)+{0,1})
    {
        int row = t >> 2, ch = (t & 3) * 2;
        unsigned d = lds_off(&s->A[buf][row][ch * 8]);
        const __bf16* g = X + (size_t)(mbase0 + row) * DIM + k + ch * 8;
        ASYNC_CP16(d, g, 0);
        ASYNC_CP16(d, g, 16);
    }
}

__device__ __forceinline__ void gemm_core(GemmLds* s,
                                          const __bf16* X, const __bf16* W,
                                          int mbase0, int nbase0,
                                          int lane, int wave, int t,
                                          v8f acc[2][4]) {
    gemm_stage(s, 0, X, W, mbase0, nbase0, 0, t);
    wait_async0();
    __syncthreads();

    int arow = (wave & 1) * 32;
    int brow = (wave >> 1) * 64;
    int cur = 0;
    for (int k = 0; k < DIM; k += 64) {
        int nxt = cur ^ 1;
        if (k + 64 < DIM)
            gemm_stage(s, nxt, X, W, mbase0, nbase0, k + 64, t);

        const __bf16* Ab = &s->A[cur][0][0];
        const __bf16* Bb = &s->B[cur][0][0];
#pragma unroll
        for (int kc = 0; kc < 64; kc += 32) {
            // batch fragment loads, then fire 8 WMMAs in 2 independent chains
            v16bf a0 = load_frag(Ab, 64, arow,      kc, lane);
            v16bf a1 = load_frag(Ab, 64, arow + 16, kc, lane);
            v16bf b0 = load_frag(Bb, 64, brow +  0, kc, lane);
            v16bf b1 = load_frag(Bb, 64, brow + 16, kc, lane);
            v16bf b2 = load_frag(Bb, 64, brow + 32, kc, lane);
            v16bf b3 = load_frag(Bb, 64, brow + 48, kc, lane);
            acc[0][0] = wmma_bf16(a0, b0, acc[0][0]);
            acc[1][0] = wmma_bf16(a1, b0, acc[1][0]);
            acc[0][1] = wmma_bf16(a0, b1, acc[0][1]);
            acc[1][1] = wmma_bf16(a1, b1, acc[1][1]);
            acc[0][2] = wmma_bf16(a0, b2, acc[0][2]);
            acc[1][2] = wmma_bf16(a1, b2, acc[1][2]);
            acc[0][3] = wmma_bf16(a0, b3, acc[0][3]);
            acc[1][3] = wmma_bf16(a1, b3, acc[1][3]);
        }
        wait_async0();
        __syncthreads();
        cur = nxt;
    }
}

// ---------------------------------------------------------------------------
// Kernel 1: QKV GEMM.  grid = (4096/64, 3072/256), block = 256.
// Scatter: Q,K -> [BH,SEQ,64] bf16; V -> transposed [BH,64,SEQ] bf16.
// ---------------------------------------------------------------------------
__global__ void qkv_gemm(const __bf16* __restrict__ X,
                         const __bf16* __restrict__ W,
                         const float*  __restrict__ bias,
                         __bf16* __restrict__ Qo,
                         __bf16* __restrict__ Ko,
                         __bf16* __restrict__ Vt) {
    __shared__ GemmLds s;
    int t = threadIdx.x, lane = t & 31, wave = t >> 5;
    int mbase0 = blockIdx.x * 64;
    int nbase0 = blockIdx.y * 256;

    v8f acc[2][4] = {};
    gemm_core(&s, X, W, mbase0, nbase0, lane, wave, t, acc);

    int kh = lane >> 4;
    int mbase = mbase0 + (wave & 1) * 32;
    int nbase = nbase0 + (wave >> 1) * 64;
#pragma unroll
    for (int r = 0; r < 2; ++r) {
#pragma unroll
        for (int f = 0; f < 4; ++f) {
#pragma unroll
            for (int v = 0; v < 8; ++v) {
                int m = mbase + r * 16 + v + (kh << 3);
                int n = nbase + f * 16 + (lane & 15);
                __bf16 bv = f2bf(acc[r][f][v] + bias[n]);
                int which = n >> 10;          // 0=q 1=k 2=v
                int d  = n & (DIM - 1);
                int h  = d >> 6;
                int hd = d & 63;
                int bb = m >> 11;
                int sidx = m & (SEQ - 1);
                int bh = bb * NHEADS + h;
                if (which == 0)
                    Qo[((size_t)bh * SEQ + sidx) * HEADD + hd] = bv;
                else if (which == 1)
                    Ko[((size_t)bh * SEQ + sidx) * HEADD + hd] = bv;
                else
                    Vt[((size_t)bh * HEADD + hd) * SEQ + sidx] = bv;
            }
        }
    }
}

// ---------------------------------------------------------------------------
// Kernel 2: flash attention per (b,h), 64-key tiles.  All 8 waves share the
// K/V tiles (async-staged, double buffered).  Each wave owns 16 query rows;
// online softmax; P routed through per-wave LDS to fix the C->A layout.
// grid = (SEQ/128, BH), block = 256.  LDS = 48 KB.
// ---------------------------------------------------------------------------
struct AttnLds {
    __align__(16) __bf16 K[2][64][64];    // 16 KB (keys x dims)
    __align__(16) __bf16 V[2][64][64];    // 16 KB (dims x keys, from Vt)
    __align__(16) __bf16 P[8][16][64];    // 16 KB (per-wave)
};

__device__ __forceinline__ void attn_stage(AttnLds* s, int buf,
                                           const __bf16* Kh,
                                           const __bf16* Vh,
                                           int kt, int t) {
    int row = t >> 2;            // 0..63
    int ch  = (t & 3) * 2;       // chunk pairs: 2 x 16 B per thread per tile
    // K tile 64 x 64 (keys x dims)
    {
        unsigned d = lds_off(&s->K[buf][row][ch * 8]);
        const __bf16* g = Kh + (size_t)(kt + row) * HEADD + ch * 8;
        ASYNC_CP16(d, g, 0);
        ASYNC_CP16(d, g, 16);
    }
    // V tile 64 x 64 (dims x keys)
    {
        unsigned d = lds_off(&s->V[buf][row][ch * 8]);
        const __bf16* g = Vh + (size_t)row * SEQ + kt + ch * 8;
        ASYNC_CP16(d, g, 0);
        ASYNC_CP16(d, g, 16);
    }
}

__global__ void attn_fwd(const __bf16* __restrict__ Qb,
                         const __bf16* __restrict__ Kb,
                         const __bf16* __restrict__ Vt,
                         __bf16* __restrict__ Ob) {
    __shared__ AttnLds s;
    int t = threadIdx.x, lane = t & 31, wave = t >> 5;
    int bh = blockIdx.y;

    const __bf16* Qh = Qb + (size_t)bh * SEQ * HEADD;
    const __bf16* Kh = Kb + (size_t)bh * SEQ * HEADD;
    const __bf16* Vh = Vt + (size_t)bh * HEADD * SEQ;

    int qbase = blockIdx.x * 128 + wave * 16;
    v16bf qa0 = load_frag(Qh, HEADD, qbase, 0, lane);
    v16bf qa1 = load_frag(Qh, HEADD, qbase, 32, lane);

    v8f acc[4] = {};
    float rm[8], rl[8];
#pragma unroll
    for (int v = 0; v < 8; ++v) { rm[v] = -1e30f; rl[v] = 0.0f; }

    const float scale = 0.125f;   // 1/sqrt(64)
    int kh = lane >> 4;

    attn_stage(&s, 0, Kh, Vh, 0, t);
    wait_async0();
    __syncthreads();

    int cur = 0;
    for (int kt = 0; kt < SEQ; kt += 64) {
        int nxt = cur ^ 1;
        if (kt + 64 < SEQ)
            attn_stage(&s, nxt, Kh, Vh, kt + 64, t);

        // --- S = Q * K^T over 64 keys: sf[j] covers keys j*16..j*16+15 ---
        const __bf16* Kt = &s.K[cur][0][0];
        v8f sf[4] = {};
#pragma unroll
        for (int kc = 0; kc < 64; kc += 32) {
            v16bf qa = (kc == 0) ? qa0 : qa1;
            v16bf k0 = load_frag(Kt, 64,  0, kc, lane);
            v16bf k1 = load_frag(Kt, 64, 16, kc, lane);
            v16bf k2 = load_frag(Kt, 64, 32, kc, lane);
            v16bf k3 = load_frag(Kt, 64, 48, kc, lane);
            sf[0] = wmma_bf16(qa, k0, sf[0]);
            sf[1] = wmma_bf16(qa, k1, sf[1]);
            sf[2] = wmma_bf16(qa, k2, sf[2]);
            sf[3] = wmma_bf16(qa, k3, sf[3]);
        }

        // --- online softmax; row m = v + 8*kh across 16-lane halves ---
#pragma unroll
        for (int v = 0; v < 8; ++v) {
            float x0 = sf[0][v] * scale;
            float x1 = sf[1][v] * scale;
            float x2 = sf[2][v] * scale;
            float x3 = sf[3][v] * scale;
            float mx = fmaxf(fmaxf(x0, x1), fmaxf(x2, x3));
            mx = fmaxf(mx, __shfl_xor(mx, 8, 32));
            mx = fmaxf(mx, __shfl_xor(mx, 4, 32));
            mx = fmaxf(mx, __shfl_xor(mx, 2, 32));
            mx = fmaxf(mx, __shfl_xor(mx, 1, 32));
            float mnew = fmaxf(rm[v], mx);
            float al   = __expf(rm[v] - mnew);
            float p0   = __expf(x0 - mnew);
            float p1   = __expf(x1 - mnew);
            float p2   = __expf(x2 - mnew);
            float p3   = __expf(x3 - mnew);
            float sum  = (p0 + p1) + (p2 + p3);
            sum += __shfl_xor(sum, 8, 32);
            sum += __shfl_xor(sum, 4, 32);
            sum += __shfl_xor(sum, 2, 32);
            sum += __shfl_xor(sum, 1, 32);
            rl[v] = rl[v] * al + sum;
            rm[v] = mnew;
            int row = v + (kh << 3);
            int c   = lane & 15;
            s.P[wave][row][c]      = f2bf(p0);
            s.P[wave][row][16 + c] = f2bf(p1);
            s.P[wave][row][32 + c] = f2bf(p2);
            s.P[wave][row][48 + c] = f2bf(p3);
#pragma unroll
            for (int f = 0; f < 4; ++f) acc[f][v] *= al;
        }

        // --- acc += P(16x64) * V(64x64) ---
        const __bf16* Pb = &s.P[wave][0][0];
        const __bf16* Vb = &s.V[cur][0][0];
#pragma unroll
        for (int kc = 0; kc < 64; kc += 32) {
            v16bf pa = load_frag(Pb, 64, 0, kc, lane);
            v16bf v0 = load_frag(Vb, 64,  0, kc, lane);
            v16bf v1 = load_frag(Vb, 64, 16, kc, lane);
            v16bf v2 = load_frag(Vb, 64, 32, kc, lane);
            v16bf v3 = load_frag(Vb, 64, 48, kc, lane);
            acc[0] = wmma_bf16(pa, v0, acc[0]);
            acc[1] = wmma_bf16(pa, v1, acc[1]);
            acc[2] = wmma_bf16(pa, v2, acc[2]);
            acc[3] = wmma_bf16(pa, v3, acc[3]);
        }

        wait_async0();
        __syncthreads();
        cur = nxt;
    }

    // --- normalize, store Ob [B, S, H*64] bf16 ---
    int bb = bh >> 4;
    int h  = bh & 15;
#pragma unroll
    for (int f = 0; f < 4; ++f) {
#pragma unroll
        for (int v = 0; v < 8; ++v) {
            int sidx = qbase + v + (kh << 3);
            int c = h * HEADD + f * 16 + (lane & 15);
            Ob[((size_t)bb * SEQ + sidx) * DIM + c] = f2bf(acc[f][v] / rl[v]);
        }
    }
}

// ---------------------------------------------------------------------------
// Kernel 3: output projection (f32 out + bias).
// grid = (4096/64, 1024/256), block = 256.
// ---------------------------------------------------------------------------
__global__ void proj_gemm(const __bf16* __restrict__ A,
                          const __bf16* __restrict__ W,
                          const float*  __restrict__ bias,
                          float* __restrict__ out) {
    __shared__ GemmLds s;
    int t = threadIdx.x, lane = t & 31, wave = t >> 5;
    int mbase0 = blockIdx.x * 64;
    int nbase0 = blockIdx.y * 256;

    v8f acc[2][4] = {};
    gemm_core(&s, A, W, mbase0, nbase0, lane, wave, t, acc);

    int kh = lane >> 4;
    int mbase = mbase0 + (wave & 1) * 32;
    int nbase = nbase0 + (wave >> 1) * 64;
#pragma unroll
    for (int r = 0; r < 2; ++r) {
#pragma unroll
        for (int f = 0; f < 4; ++f) {
#pragma unroll
            for (int v = 0; v < 8; ++v) {
                int m = mbase + r * 16 + v + (kh << 3);
                int n = nbase + f * 16 + (lane & 15);
                out[(size_t)m * DIM + n] = acc[r][f][v] + bias[n];
            }
        }
    }
}

// ---------------------------------------------------------------------------
extern "C" void kernel_launch(void* const* d_in, const int* in_sizes, int n_in,
                              void* d_out, int out_size, void* d_ws, size_t ws_size,
                              hipStream_t stream) {
    const float* x      = (const float*)d_in[0];
    const float* w_qkv  = (const float*)d_in[1];
    const float* b_qkv  = (const float*)d_in[2];
    const float* w_proj = (const float*)d_in[3];
    const float* b_proj = (const float*)d_in[4];
    float* out = (float*)d_out;

    char* ws = (char*)d_ws;
    size_t off = 0;
    __bf16* Xb  = (__bf16*)(ws + off); off += (size_t)MROWS * DIM * 2;
    __bf16* Wqb = (__bf16*)(ws + off); off += (size_t)3 * DIM * DIM * 2;
    __bf16* Wpb = (__bf16*)(ws + off); off += (size_t)DIM * DIM * 2;
    __bf16* Qb  = (__bf16*)(ws + off); off += (size_t)BH * SEQ * HEADD * 2;
    __bf16* Kb  = (__bf16*)(ws + off); off += (size_t)BH * SEQ * HEADD * 2;
    __bf16* Vt  = (__bf16*)(ws + off); off += (size_t)BH * HEADD * SEQ * 2;
    __bf16* Ob  = (__bf16*)(ws + off); off += (size_t)MROWS * DIM * 2;

    int nx = MROWS * DIM;
    int nw = 3 * DIM * DIM;
    int np = DIM * DIM;
    cvt_f32_to_bf16<<<(nx + 255) / 256, 256, 0, stream>>>(x, Xb, nx);
    cvt_f32_to_bf16<<<(nw + 255) / 256, 256, 0, stream>>>(w_qkv, Wqb, nw);
    cvt_f32_to_bf16<<<(np + 255) / 256, 256, 0, stream>>>(w_proj, Wpb, np);

    qkv_gemm<<<dim3(MROWS / 64, (3 * DIM) / 256), 256, 0, stream>>>(
        Xb, Wqb, b_qkv, Qb, Kb, Vt);

    attn_fwd<<<dim3(SEQ / 128, BH), 256, 0, stream>>>(Qb, Kb, Vt, Ob);

    proj_gemm<<<dim3(MROWS / 64, DIM / 256), 256, 0, stream>>>(
        Ob, Wpb, b_proj, out);
}
